// Decoder_70325794504898
// MI455X (gfx1250) — matrix operand order repaired and checked by reading, not compiled, not run on hardware
//
#include <hip/hip_runtime.h>
#include <hip/hip_bf16.h>

// ---------------- problem constants ----------------
#define BB 32
#define TT 64
#define SS 64
#define VV 32000
#define EE 512
#define HH 512
#define LL 2
#define G4H 2048            // 4*H
#define K0  1024            // E + H  (layer0 input width)
#define KC  1536            // 2H + E (fc concat width)

typedef __bf16 bf16_t;
typedef __attribute__((ext_vector_type(16))) __bf16 v16bf;
typedef __attribute__((ext_vector_type(8)))  float  v8f;

// ---------------- WMMA fragment loaders ----------------
// A fragment: 16x32 bf16, M x K, row-major source with leading dim lda.
// Lane layout (ISA 7.12.2): lanes 0-15 -> M=lane, K = {0..7, 16..23};
//                           lanes 16-31 -> M=lane-16, K = {8..15, 24..31}.
__device__ __forceinline__ v16bf load_a_frag(const bf16_t* __restrict__ X, int lda,
                                             int m0, int k0, int lane) {
    int row  = m0 + (lane & 15);
    int off  = 8 * (lane >> 4);
    const bf16_t* p = X + (long)row * lda + k0 + off;
    union { v16bf v; uint4 u[2]; } a;
    a.u[0] = *(const uint4*)(p);        // K = k0+off    .. +7
    a.u[1] = *(const uint4*)(p + 16);   // K = k0+off+16 .. +23
    return a.v;
}

// B fragment: 32x16 bf16, K x N, k-major source (row-major K x N, leading dim N).
// Lane layout: N = lane & 15; lanes 0-15 -> K = {0..7, 16..23}, lanes 16-31 -> K = {8..15, 24..31}.
__device__ __forceinline__ v16bf load_b_frag(const bf16_t* __restrict__ Wt, int N,
                                             int k0, int n0, int lane) {
    const bf16_t* p = Wt + (long)(k0 + 8 * (lane >> 4)) * N + n0 + (lane & 15);
    v16bf b;
#pragma unroll
    for (int s = 0; s < 8; s++) {
        b[s]     = p[(long)s * N];
        b[8 + s] = p[(long)(16 + s) * N];
    }
    return b;
}

#define WMMA_BF16(A, Bm, C) \
    __builtin_amdgcn_wmma_f32_16x16x32_bf16(false, (A), false, (Bm), (short)0, (C), false, false)

// ---------------- weight conversion: f32 (R x C) -> bf16 transposed (C x R) ----------------
__global__ void k_convT(const float* __restrict__ in, bf16_t* __restrict__ out,
                        int R, int C) {
    long total = (long)R * C;
    for (long i = (long)blockIdx.x * blockDim.x + threadIdx.x; i < total;
         i += (long)gridDim.x * blockDim.x) {
        long r = i / C, c = i % C;
        out[c * R + r] = (bf16_t)in[i];
    }
}

__global__ void k_f2bf(const float* __restrict__ in, bf16_t* __restrict__ out, int n) {
    int i = blockIdx.x * blockDim.x + threadIdx.x;
    if (i < n) out[i] = (bf16_t)in[i];
}

// ---------------- attention + embedding (one block per batch row) ----------------
__global__ void k_attn(int t, const int* __restrict__ tok, const float* __restrict__ emb,
                       const float* __restrict__ enc, const float* __restrict__ hstate,
                       bf16_t* __restrict__ xbuf, bf16_t* __restrict__ concat,
                       float* __restrict__ attn_out) {
    int b   = blockIdx.x;
    int tid = threadIdx.x;
    __shared__ float part[SS][4];
    __shared__ float sc[SS];

    const float* q  = hstate + (long)(LL - 1) * BB * HH + (long)b * HH;  // h[-1][b]
    const float* eb = enc + (long)b * SS * HH;

    // scores: 4 partial sums per source position
    int s = tid >> 2, p4 = tid & 3;
    float acc = 0.f;
    for (int h = p4; h < HH; h += 4) acc += q[h] * eb[(long)s * HH + h];
    part[s][p4] = acc;
    __syncthreads();
    if (tid < SS)
        sc[tid] = (part[tid][0] + part[tid][1] + part[tid][2] + part[tid][3])
                  * 0.04419417382415922f;   // 1/sqrt(512)
    __syncthreads();
    if (tid == 0) {
        float m = sc[0];
        for (int i = 1; i < SS; i++) m = fmaxf(m, sc[i]);
        float sum = 0.f;
        for (int i = 0; i < SS; i++) { float e = __expf(sc[i] - m); sc[i] = e; sum += e; }
        float inv = 1.f / sum;
        for (int i = 0; i < SS; i++) sc[i] *= inv;
    }
    __syncthreads();
    if (tid < SS) attn_out[((long)b * TT + t) * SS + tid] = sc[tid];

    int token = tok[b * TT + t];
    const float* er = emb + (long)token * EE;
    long crow = ((long)t * BB + b) * KC;

    // context vector + write bf16 activations: x = [emb, ctx], concat = [h1, ctx, emb]
    for (int hh = tid; hh < HH; hh += blockDim.x) {
        float cx = 0.f;
        for (int ss = 0; ss < SS; ss++) cx += sc[ss] * eb[(long)ss * HH + hh];
        xbuf[b * K0 + EE + hh]  = (bf16_t)cx;
        concat[crow + HH + hh]  = (bf16_t)cx;
    }
    for (int e = tid; e < EE; e += blockDim.x) {
        float v = er[e];
        xbuf[b * K0 + e]            = (bf16_t)v;
        concat[crow + 2 * HH + e]   = (bf16_t)v;
    }
}

// ---------------- LSTM gates GEMM: G(32 x 2048) = Xa*Wa^T + Xb*Wb^T (WMMA bf16) -----
// Wa/Wb pre-transposed to k-major (Ka x 2048) / (Kb x 2048).
// 256 wave-tiles of 16x16; 8 waves per block, 32 blocks.
__global__ void k_gemm_gates(const bf16_t* __restrict__ Xa, int Ka,
                             const bf16_t* __restrict__ Wat,
                             const bf16_t* __restrict__ Xb, int Kb,
                             const bf16_t* __restrict__ Wbt,
                             float* __restrict__ Gout) {
    int lane = threadIdx.x & 31;
    int wave = threadIdx.x >> 5;
    int tile = blockIdx.x * 8 + wave;
    int m0 = (tile & 1) * 16;
    int n0 = (tile >> 1) * 16;
    v8f acc = {};
    for (int k = 0; k < Ka; k += 32) {
        v16bf a = load_a_frag(Xa, Ka, m0, k, lane);
        v16bf b = load_b_frag(Wat, G4H, k, n0, lane);
        acc = WMMA_BF16(a, b, acc);
    }
    for (int k = 0; k < Kb; k += 32) {
        v16bf a = load_a_frag(Xb, Kb, m0, k, lane);
        v16bf b = load_b_frag(Wbt, G4H, k, n0, lane);
        acc = WMMA_BF16(a, b, acc);
    }
    int half = lane >> 4, col = lane & 15;
#pragma unroll
    for (int r = 0; r < 8; r++) {
        int m = m0 + r + 8 * half;
        Gout[(long)m * G4H + n0 + col] = acc[r];
    }
}

// ---------------- LSTM elementwise (PyTorch gate order i,f,g,o) ----------------
__global__ void k_lstm_ew(const float* __restrict__ G, const float* __restrict__ b_ih,
                          const float* __restrict__ b_hh, float* __restrict__ hst,
                          float* __restrict__ cst, bf16_t* __restrict__ hbf,
                          bf16_t* __restrict__ concat, int t) {
    int i = blockIdx.x * blockDim.x + threadIdx.x;
    if (i >= BB * HH) return;
    int b = i / HH, h = i % HH;
    const float* g = G + (long)b * G4H;
    float gi = g[h]          + b_ih[h]          + b_hh[h];
    float gf = g[HH + h]     + b_ih[HH + h]     + b_hh[HH + h];
    float gg = g[2 * HH + h] + b_ih[2 * HH + h] + b_hh[2 * HH + h];
    float go = g[3 * HH + h] + b_ih[3 * HH + h] + b_hh[3 * HH + h];
    float si = 1.f / (1.f + __expf(-gi));
    float sf = 1.f / (1.f + __expf(-gf));
    float so = 1.f / (1.f + __expf(-go));
    float c  = sf * cst[i] + si * tanhf(gg);
    float hn = so * tanhf(c);
    cst[i] = c;
    hst[i] = hn;
    hbf[i] = (bf16_t)hn;
    if (concat) concat[((long)t * BB + b) * KC + h] = (bf16_t)hn;   // concat[:, 0:H] = h1
}

// ---------------- fc projection: (2048 x 32000) = concat(2048x1536) x fcW^T ----------
// Each wave: 32x64 output tile (2 A frags x 4 B frags -> 8 WMMA per K-step).
// Doubles L2 arithmetic intensity vs 16x64 and issues 8 independent WMMAs
// back-to-back to hide XDL latency. K = 1536 -> 48 K-steps -> 384 v_wmma/wave.
__global__ void k_fc_gemm(const bf16_t* __restrict__ Cc, const bf16_t* __restrict__ Wt,
                          const float* __restrict__ bias, float* __restrict__ out) {
    int lane = threadIdx.x & 31;
    int wave = threadIdx.x >> 5;
    int m0 = blockIdx.y * 32;
    int n0 = (blockIdx.x * 8 + wave) * 64;
    if (n0 >= VV) return;
    v8f c00 = {}, c01 = {}, c02 = {}, c03 = {};
    v8f c10 = {}, c11 = {}, c12 = {}, c13 = {};
    for (int k = 0; k < KC; k += 32) {
        // prefetch next K panel of B into cache hierarchy (global_prefetch_b8)
        if (k + 32 < KC)
            __builtin_prefetch(Wt + (long)(k + 32 + lane) * VV + n0, 0, 1);
        v16bf a0 = load_a_frag(Cc, KC, m0,      k, lane);
        v16bf a1 = load_a_frag(Cc, KC, m0 + 16, k, lane);
        v16bf b0 = load_b_frag(Wt, VV, k, n0,      lane);
        v16bf b1 = load_b_frag(Wt, VV, k, n0 + 16, lane);
        v16bf b2 = load_b_frag(Wt, VV, k, n0 + 32, lane);
        v16bf b3 = load_b_frag(Wt, VV, k, n0 + 48, lane);
        c00 = WMMA_BF16(a0, b0, c00);
        c01 = WMMA_BF16(a0, b1, c01);
        c02 = WMMA_BF16(a0, b2, c02);
        c03 = WMMA_BF16(a0, b3, c03);
        c10 = WMMA_BF16(a1, b0, c10);
        c11 = WMMA_BF16(a1, b1, c11);
        c12 = WMMA_BF16(a1, b2, c12);
        c13 = WMMA_BF16(a1, b3, c13);
    }
    int half = lane >> 4, col = lane & 15;
#pragma unroll
    for (int r = 0; r < 8; r++) {
        {   // M sub-tile 0
            int m  = m0 + r + 8 * half;     // GEMM row = t*B + b
            int tt = m >> 5;                // / B
            int bb = m & 31;                // % B
            float* o = out + ((long)bb * TT + tt) * VV;   // output layout (B, T, V)
            o[n0 + col]      = c00[r] + bias[n0 + col];
            o[n0 + 16 + col] = c01[r] + bias[n0 + 16 + col];
            o[n0 + 32 + col] = c02[r] + bias[n0 + 32 + col];
            o[n0 + 48 + col] = c03[r] + bias[n0 + 48 + col];
        }
        {   // M sub-tile 1
            int m  = m0 + 16 + r + 8 * half;
            int tt = m >> 5;
            int bb = m & 31;
            float* o = out + ((long)bb * TT + tt) * VV;
            o[n0 + col]      = c10[r] + bias[n0 + col];
            o[n0 + 16 + col] = c11[r] + bias[n0 + 16 + col];
            o[n0 + 32 + col] = c12[r] + bias[n0 + 32 + col];
            o[n0 + 48 + col] = c13[r] + bias[n0 + 48 + col];
        }
    }
}

// ---------------- host launch ----------------
extern "C" void kernel_launch(void* const* d_in, const int* in_sizes, int n_in,
                              void* d_out, int out_size, void* d_ws, size_t ws_size,
                              hipStream_t stream) {
    const int*   ti   = (const int*)  d_in[0];
    const float* enc  = (const float*)d_in[2];
    const float* h0   = (const float*)d_in[3];
    const float* c0   = (const float*)d_in[4];
    const float* emb  = (const float*)d_in[6];
    const float* Wih0 = (const float*)d_in[7];
    const float* Whh0 = (const float*)d_in[8];
    const float* bih0 = (const float*)d_in[9];
    const float* bhh0 = (const float*)d_in[10];
    const float* Wih1 = (const float*)d_in[11];
    const float* Whh1 = (const float*)d_in[12];
    const float* bih1 = (const float*)d_in[13];
    const float* bhh1 = (const float*)d_in[14];
    const float* fcW  = (const float*)d_in[15];
    const float* fcb  = (const float*)d_in[16];

    float* out_logits = (float*)d_out;                               // (B,T,V)
    float* out_h      = out_logits + (long)BB * TT * VV;             // (L,B,H)
    float* out_c      = out_h + (long)LL * BB * HH;                  // (L,B,H)
    float* out_attn   = out_c + (long)LL * BB * HH;                  // (B,T,S)

    char* p = (char*)d_ws;
    auto alloc = [&](size_t bytes) -> void* {
        void* r = (void*)p;
        p += (bytes + 255) & ~(size_t)255;
        return r;
    };
    bf16_t* Wih0t  = (bf16_t*)alloc((size_t)K0 * G4H * 2);
    bf16_t* Whh0t  = (bf16_t*)alloc((size_t)HH * G4H * 2);
    bf16_t* Wih1t  = (bf16_t*)alloc((size_t)HH * G4H * 2);
    bf16_t* Whh1t  = (bf16_t*)alloc((size_t)HH * G4H * 2);
    bf16_t* fcWt   = (bf16_t*)alloc((size_t)KC * VV * 2);
    bf16_t* concat = (bf16_t*)alloc((size_t)BB * TT * KC * 2);
    bf16_t* xbuf   = (bf16_t*)alloc((size_t)BB * K0 * 2);
    bf16_t* hbf    = (bf16_t*)alloc((size_t)LL * BB * HH * 2);
    float*  hst    = (float*) alloc((size_t)LL * BB * HH * 4);
    float*  cst    = (float*) alloc((size_t)LL * BB * HH * 4);
    float*  Gbuf   = (float*) alloc((size_t)BB * G4H * 4);

    // recurrent state (do not mutate inputs)
    hipMemcpyAsync(hst, h0, (size_t)LL * BB * HH * 4, hipMemcpyDeviceToDevice, stream);
    hipMemcpyAsync(cst, c0, (size_t)LL * BB * HH * 4, hipMemcpyDeviceToDevice, stream);
    k_f2bf<<<(LL * BB * HH + 255) / 256, 256, 0, stream>>>(hst, hbf, LL * BB * HH);

    // one-time weight conversion to bf16, transposed to k-major for B fragments
    k_convT<<<512, 256, 0, stream>>>(Wih0, Wih0t, G4H, K0);
    k_convT<<<512, 256, 0, stream>>>(Whh0, Whh0t, G4H, HH);
    k_convT<<<512, 256, 0, stream>>>(Wih1, Wih1t, G4H, HH);
    k_convT<<<512, 256, 0, stream>>>(Whh1, Whh1t, G4H, HH);
    k_convT<<<4096, 256, 0, stream>>>(fcW, fcWt, VV, KC);

    // sequential decode scan
    for (int t = 0; t < TT; t++) {
        k_attn<<<BB, 256, 0, stream>>>(t, ti, emb, enc, hst, xbuf, concat, out_attn);
        // layer 0: x = [emb, ctx] (K=1024) and h0_state (K=512)
        k_gemm_gates<<<32, 256, 0, stream>>>(xbuf, K0, Wih0t, hbf, HH, Whh0t, Gbuf);
        k_lstm_ew<<<(BB * HH + 255) / 256, 256, 0, stream>>>(
            Gbuf, bih0, bhh0, hst, cst, hbf, (bf16_t*)nullptr, t);
        // layer 1: x = h_layer0_new, h = h_layer1_state
        k_gemm_gates<<<32, 256, 0, stream>>>(hbf, HH, Wih1t, hbf + BB * HH, HH, Whh1t, Gbuf);
        k_lstm_ew<<<(BB * HH + 255) / 256, 256, 0, stream>>>(
            Gbuf, bih1, bhh1, hst + BB * HH, cst + BB * HH, hbf + BB * HH, concat, t);
    }

    // batched vocabulary projection over all (t, b) rows: 2048 x 32000 x 1536
    dim3 fg(63, 64);    // 63*8 waves * 64 cols >= 32000 ; 64 M-tiles of 32
    k_fc_gemm<<<fg, 256, 0, stream>>>(concat, fcWt, fcb, out_logits);

    // final states
    hipMemcpyAsync(out_h, hst, (size_t)LL * BB * HH * 4, hipMemcpyDeviceToDevice, stream);
    hipMemcpyAsync(out_c, cst, (size_t)LL * BB * HH * 4, hipMemcpyDeviceToDevice, stream);
}
